// SAGE_23175643530075
// MI455X (gfx1250) — compile-verified, matrix operand order checked by reading, not run on hardware
//
#include <hip/hip_runtime.h>
#include <hip/hip_bf16.h>

typedef __attribute__((ext_vector_type(16))) _Float16 v16h;
typedef __attribute__((ext_vector_type(8)))  _Float16 v8h;
typedef __attribute__((ext_vector_type(8)))  float    v8f;

#define N_NODES   100000
#define N_EDGES   1600000
#define FEATS     128          // K dim for every layer (in=128, hidden=128)
#define TPB       256

// ---------------------------------------------------------------- utilities
__global__ void zero_f4_kernel(float4* __restrict__ p, int n4) {
    int i = blockIdx.x * blockDim.x + threadIdx.x;
    if (i < n4) p[i] = make_float4(0.f, 0.f, 0.f, 0.f);
}

__global__ void degree_kernel(const int* __restrict__ dst, float* __restrict__ deg, int n_edges) {
    int i = blockIdx.x * blockDim.x + threadIdx.x;
    if (i < n_edges) atomicAdd(&deg[dst[i]], 1.0f);
}

__global__ void invdeg_kernel(const float* __restrict__ deg, float* __restrict__ invd, int n) {
    int i = blockIdx.x * blockDim.x + threadIdx.x;
    if (i < n) invd[i] = 1.0f / fmaxf(deg[i], 1.0f);
}

// Transpose W[K,N] (row-major fp32) -> WT[N,K] fp16 (column-major for the GEMM).
__global__ void wt_f16_kernel(const float* __restrict__ W, _Float16* __restrict__ WT,
                              int K, int N) {
    int i = blockIdx.x * blockDim.x + threadIdx.x;   // over N*K, WT-linear
    if (i >= K * N) return;
    int n = i / K;
    int k = i - n * K;
    WT[i] = (_Float16)W[(size_t)k * N + n];
}

// ------------------------------------------------------- edge scatter (sum)
// One wave32 per edge: 32 lanes x float4 = 128 features, 512B coalesced gather.
__global__ void scatter_add_kernel(const float* __restrict__ H,
                                   const int* __restrict__ src,
                                   const int* __restrict__ dst,
                                   float* __restrict__ agg,
                                   int n_edges) {
    int gtid = blockIdx.x * blockDim.x + threadIdx.x;
    int edge = gtid >> 5;
    int lane = threadIdx.x & 31;
    if (edge >= n_edges) return;
    int s = src[edge];
    int d = dst[edge];
    const float4* hp = (const float4*)(H + (size_t)s * FEATS);
    float4 v = hp[lane];
    float* ap = agg + (size_t)d * FEATS + lane * 4;
    atomicAdd(ap + 0, v.x);
    atomicAdd(ap + 1, v.y);
    atomicAdd(ap + 2, v.z);
    atomicAdd(ap + 3, v.w);
}

// ------------------------------------------------- fused SAGE layer (WMMA)
// out[i,:] = act( H[i,:] @ Ws + (agg[i,:]*invd[i]) @ Wn + b )
// One wave32 computes a 16x64 output tile: 4 column tiles sharing the A
// fragments; K=128 -> 4 k-chunks; 2 WMMAs (self+neigh) per (chunk, col tile)
// => 32 v_wmma_f32_16x16x32_f16 per wave, fp32 accumulation.
//
// Fragment layout facts used (CDNA5 ISA):
//  * A 16x32 f16: lane m = lane&15, half = lane>>4; v16h elements [0..7] are
//    contiguous K = kb+half*8 .. +7, elements [8..15] are K = kb+16+half*8 .. +7.
//  * B 32x16 f16 column-major mirrors A with the lane owning column n.
//  * C/D v8f: row = r + 8*half, col = lane&15.
__global__ void sage_layer_wmma_kernel(const float* __restrict__ H,
                                       const float* __restrict__ AGG,
                                       const float* __restrict__ INVD,
                                       const _Float16* __restrict__ WsT, // [n_out,128] f16
                                       const _Float16* __restrict__ WnT, // [n_out,128] f16
                                       const float* __restrict__ bias,
                                       float* __restrict__ OUT,
                                       int n_nodes, int n_out, int relu) {
    const int gtid    = blockIdx.x * blockDim.x + threadIdx.x;
    const int wave    = gtid >> 5;
    const int lane    = threadIdx.x & 31;
    const int rtiles  = n_nodes >> 4;
    const int cgroups = n_out >> 6;                 // 64-wide column groups
    if (wave >= rtiles * cgroups) return;           // wave-uniform: EXEC all-1s at WMMA

    const int rt   = wave / cgroups;
    const int cg   = wave - rt * cgroups;
    const int lm   = lane & 15;
    const int half = lane >> 4;

    const int    arow = (rt << 4) + lm;
    const float  invd = INVD[arow];
    const float* hrow = H   + (size_t)arow * FEATS;
    const float* grow = AGG + (size_t)arow * FEATS;

    v8f c0 = {}, c1 = {}, c2 = {}, c3 = {};
    v8f* acc[4] = { &c0, &c1, &c2, &c3 };

    #pragma unroll
    for (int kc = 0; kc < 4; ++kc) {
        const int kb = kc << 5;                     // 32-wide K chunk
        // ---- A fragments: two contiguous 8-float chunks per fragment
        const float4* hpl = (const float4*)(hrow + kb + (half << 3));
        const float4* hph = (const float4*)(hrow + kb + 16 + (half << 3));
        const float4* gpl = (const float4*)(grow + kb + (half << 3));
        const float4* gph = (const float4*)(grow + kb + 16 + (half << 3));
        float4 s0 = hpl[0], s1 = hpl[1], s2 = hph[0], s3 = hph[1];
        float4 g0 = gpl[0], g1 = gpl[1], g2 = gph[0], g3 = gph[1];
        v16h aS, aN;
        const float* sf = (const float*)&s0;        // s0..s3 contiguous? build explicitly:
        aS[0]=(_Float16)s0.x; aS[1]=(_Float16)s0.y; aS[2]=(_Float16)s0.z; aS[3]=(_Float16)s0.w;
        aS[4]=(_Float16)s1.x; aS[5]=(_Float16)s1.y; aS[6]=(_Float16)s1.z; aS[7]=(_Float16)s1.w;
        aS[8]=(_Float16)s2.x; aS[9]=(_Float16)s2.y; aS[10]=(_Float16)s2.z; aS[11]=(_Float16)s2.w;
        aS[12]=(_Float16)s3.x; aS[13]=(_Float16)s3.y; aS[14]=(_Float16)s3.z; aS[15]=(_Float16)s3.w;
        aN[0]=(_Float16)(g0.x*invd); aN[1]=(_Float16)(g0.y*invd); aN[2]=(_Float16)(g0.z*invd); aN[3]=(_Float16)(g0.w*invd);
        aN[4]=(_Float16)(g1.x*invd); aN[5]=(_Float16)(g1.y*invd); aN[6]=(_Float16)(g1.z*invd); aN[7]=(_Float16)(g1.w*invd);
        aN[8]=(_Float16)(g2.x*invd); aN[9]=(_Float16)(g2.y*invd); aN[10]=(_Float16)(g2.z*invd); aN[11]=(_Float16)(g2.w*invd);
        aN[12]=(_Float16)(g3.x*invd); aN[13]=(_Float16)(g3.y*invd); aN[14]=(_Float16)(g3.z*invd); aN[15]=(_Float16)(g3.w*invd);
        (void)sf;

        // ---- 4 column tiles reuse aS/aN
        #pragma unroll
        for (int t = 0; t < 4; ++t) {
            const int col = (cg << 6) + (t << 4) + lm;
            const _Float16* ws = WsT + (size_t)col * FEATS + kb + (half << 3);
            const _Float16* wn = WnT + (size_t)col * FEATS + kb + (half << 3);
            v8h bs_lo = *(const v8h*)(ws);
            v8h bs_hi = *(const v8h*)(ws + 16);
            v8h bn_lo = *(const v8h*)(wn);
            v8h bn_hi = *(const v8h*)(wn + 16);
            v16h bS = __builtin_shufflevector(bs_lo, bs_hi,
                        0,1,2,3,4,5,6,7,8,9,10,11,12,13,14,15);
            v16h bN = __builtin_shufflevector(bn_lo, bn_hi,
                        0,1,2,3,4,5,6,7,8,9,10,11,12,13,14,15);
            *acc[t] = __builtin_amdgcn_wmma_f32_16x16x32_f16(
                        false, aS, false, bS, (short)0, *acc[t], false, false);
            *acc[t] = __builtin_amdgcn_wmma_f32_16x16x32_f16(
                        false, aN, false, bN, (short)0, *acc[t], false, false);
        }
    }

    // ---- epilogue: bias + optional ReLU
    #pragma unroll
    for (int t = 0; t < 4; ++t) {
        const int col = (cg << 6) + (t << 4) + lm;
        const float bv = bias[col];
        #pragma unroll
        for (int r = 0; r < 8; ++r) {
            const int orow = (rt << 4) + r + (half << 3);
            float v = (*acc[t])[r] + bv;
            if (relu) v = fmaxf(v, 0.0f);
            OUT[(size_t)orow * n_out + col] = v;
        }
    }
}

// --------------------------------------------------------------- launcher
static inline void zero_buf(float* p, size_t nfloats, hipStream_t stream) {
    int n4 = (int)(nfloats >> 2);
    zero_f4_kernel<<<(n4 + TPB - 1) / TPB, TPB, 0, stream>>>((float4*)p, n4);
}

extern "C" void kernel_launch(void* const* d_in, const int* in_sizes, int n_in,
                              void* d_out, int out_size, void* d_ws, size_t ws_size,
                              hipStream_t stream) {
    const float* x   = (const float*)d_in[0];
    const int*   src = (const int*)  d_in[1];
    const int*   dst = (const int*)  d_in[2];
    const float* Ws1 = (const float*)d_in[3];
    const float* Wn1 = (const float*)d_in[4];
    const float* b1  = (const float*)d_in[5];
    const float* Ws2 = (const float*)d_in[6];
    const float* Wn2 = (const float*)d_in[7];
    const float* b2  = (const float*)d_in[8];
    const float* Ws3 = (const float*)d_in[9];
    const float* Wn3 = (const float*)d_in[10];
    const float* b3  = (const float*)d_in[11];
    float* out = (float*)d_out;

    // workspace layout (floats then f16 weight transposes; base 16B aligned)
    float* deg  = (float*)d_ws;                        // N
    float* invd = deg  + N_NODES;                      // N
    float* agg  = invd + N_NODES;                      // N*128 (reused per layer)
    float* h1   = agg  + (size_t)N_NODES * FEATS;      // N*128
    float* h2   = h1   + (size_t)N_NODES * FEATS;      // N*128
    _Float16* wt = (_Float16*)(h2 + (size_t)N_NODES * FEATS);
    _Float16* Ws1T = wt;                 // 128*128
    _Float16* Wn1T = Ws1T + 128 * 128;
    _Float16* Ws2T = Wn1T + 128 * 128;
    _Float16* Wn2T = Ws2T + 128 * 128;
    _Float16* Ws3T = Wn2T + 128 * 128;   // 64*128
    _Float16* Wn3T = Ws3T + 64 * 128;

    const int edge_blocks = (N_EDGES + TPB - 1) / TPB;
    const int scat_blocks = (int)(((size_t)N_EDGES * 32 + TPB - 1) / TPB);
    const int node_blocks = (N_NODES + TPB - 1) / TPB;

    // degrees (shared by all 3 layers)
    zero_buf(deg, N_NODES, stream);
    degree_kernel<<<edge_blocks, TPB, 0, stream>>>(dst, deg, N_EDGES);
    invdeg_kernel<<<node_blocks, TPB, 0, stream>>>(deg, invd, N_NODES);

    // weight prep: transpose + fp16 convert
    wt_f16_kernel<<<(128 * 128 + TPB - 1) / TPB, TPB, 0, stream>>>(Ws1, Ws1T, 128, 128);
    wt_f16_kernel<<<(128 * 128 + TPB - 1) / TPB, TPB, 0, stream>>>(Wn1, Wn1T, 128, 128);
    wt_f16_kernel<<<(128 * 128 + TPB - 1) / TPB, TPB, 0, stream>>>(Ws2, Ws2T, 128, 128);
    wt_f16_kernel<<<(128 * 128 + TPB - 1) / TPB, TPB, 0, stream>>>(Wn2, Wn2T, 128, 128);
    wt_f16_kernel<<<(128 * 64 + TPB - 1) / TPB, TPB, 0, stream>>>(Ws3, Ws3T, 128, 64);
    wt_f16_kernel<<<(128 * 64 + TPB - 1) / TPB, TPB, 0, stream>>>(Wn3, Wn3T, 128, 64);

    const size_t aggN = (size_t)N_NODES * FEATS;

    // GEMM grids: one wave per 16x64 output tile
    const int waves_128 = (N_NODES / 16) * (128 / 64);     // 12500
    const int blocks_128 = (int)(((size_t)waves_128 * 32 + TPB - 1) / TPB);
    const int waves_64  = (N_NODES / 16) * (64 / 64);      // 6250
    const int blocks_64 = (int)(((size_t)waves_64 * 32 + TPB - 1) / TPB);

    // ---- layer 1: x -> h1
    zero_buf(agg, aggN, stream);
    scatter_add_kernel<<<scat_blocks, TPB, 0, stream>>>(x, src, dst, agg, N_EDGES);
    sage_layer_wmma_kernel<<<blocks_128, TPB, 0, stream>>>(
        x, agg, invd, Ws1T, Wn1T, b1, h1, N_NODES, 128, 1);

    // ---- layer 2: h1 -> h2
    zero_buf(agg, aggN, stream);
    scatter_add_kernel<<<scat_blocks, TPB, 0, stream>>>(h1, src, dst, agg, N_EDGES);
    sage_layer_wmma_kernel<<<blocks_128, TPB, 0, stream>>>(
        h1, agg, invd, Ws2T, Wn2T, b2, h2, N_NODES, 128, 1);

    // ---- layer 3: h2 -> out (no relu)
    zero_buf(agg, aggN, stream);
    scatter_add_kernel<<<scat_blocks, TPB, 0, stream>>>(h2, src, dst, agg, N_EDGES);
    sage_layer_wmma_kernel<<<blocks_64, TPB, 0, stream>>>(
        h2, agg, invd, Ws3T, Wn3T, b3, out, N_NODES, 64, 0);
}
